// SpectralMoEHeads_11879879540757
// MI455X (gfx1250) — matrix operand order repaired and checked by reading, not compiled, not run on hardware
//
#include <hip/hip_runtime.h>
#include <hip/hip_bf16.h>

// ---- problem constants (from reference) ----
#define B_    4
#define T_    2048
#define H_    16
#define K_    8
#define D_IN  1024
#define D_H   64
#define D_O   64
#define TTILE 32          // tokens per block in main kernel

typedef __attribute__((ext_vector_type(16))) __bf16 v16bf;
typedef __attribute__((ext_vector_type(8)))  float  v8f;

__device__ __forceinline__ unsigned short f32_to_bf16_rne(float f) {
    union { float f; unsigned int u; } c; c.f = f;
    unsigned int u = c.u;
    unsigned int r = (u + 0x7FFFu + ((u >> 16) & 1u)) >> 16;  // round-nearest-even
    return (unsigned short)r;
}

// -----------------------------------------------------------------------------
// Kernel 0: convert weight[h][k][i][o] (f32) -> bf16, pre-swizzled into the
// wave32 B-fragment layout for V_WMMA_F32_16X16X32_BF16:
//   B is 32x16 (KxN); element e of lane half hf holds
//   K = (e<8 ? 8*hf+e : 16+8*hf+(e-8)) (matches the A fragment order).
// Layout: wf[((h*8+k)*8 + chunk*4 + colTile)*512 + lane*16 + e]
//   with i = chunk*32 + (e<8 ? 0 : 16) + (lane>=16 ? 8 : 0) + (e&7),  o = colTile*16 + (lane&15)
// Each lane of the main kernel then loads its fragment with 2x b128 loads.
// -----------------------------------------------------------------------------
__global__ void prep_weight_frag_kernel(const float* __restrict__ w,
                                        unsigned short* __restrict__ wf) {
    int idx = blockIdx.x * blockDim.x + threadIdx.x;   // 0 .. 524287
    int e  =  idx        & 15;
    int l  = (idx >> 4)  & 31;
    int ct = (idx >> 9)  & 3;
    int c  = (idx >> 11) & 1;
    int k  = (idx >> 12) & 7;
    int h  = (idx >> 15) & 15;
    int i = c * 32 + ((e < 8) ? 0 : 16) + ((l >= 16) ? 8 : 0) + (e & 7);
    int o = ct * 16 + (l & 15);
    float v = w[(((size_t)(h * K_ + k)) * D_H + i) * D_O + o];
    wf[idx] = f32_to_bf16_rne(v);
}

// -----------------------------------------------------------------------------
// Kernel 1: scores[b,t,k] = phi[t,k] + sum_i x[b,t,i]*diag[k,i]
// One wave32 per token; lane-parallel over i, butterfly reduction.
// -----------------------------------------------------------------------------
__global__ void scores_kernel(const float* __restrict__ x,
                              const float* __restrict__ diag,
                              const float* __restrict__ phi,
                              float* __restrict__ scores) {
    int wave = (int)((blockIdx.x * blockDim.x + threadIdx.x) >> 5);  // token id b*T + t
    int lane = threadIdx.x & 31;
    if (wave >= B_ * T_) return;
    const float* xr = x + (size_t)wave * D_IN;

    float acc[K_];
#pragma unroll
    for (int k = 0; k < K_; ++k) acc[k] = 0.0f;

    for (int i0 = 0; i0 < D_IN; i0 += 32) {
        float xv = xr[i0 + lane];
#pragma unroll
        for (int k = 0; k < K_; ++k)
            acc[k] = fmaf(xv, diag[k * D_IN + i0 + lane], acc[k]);
    }
#pragma unroll
    for (int k = 0; k < K_; ++k) {
#pragma unroll
        for (int off = 16; off > 0; off >>= 1)
            acc[k] += __shfl_xor(acc[k], off, 32);
    }
    if (lane == 0) {
        int t = wave & (T_ - 1);
#pragma unroll
        for (int k = 0; k < K_; ++k)
            scores[(size_t)wave * K_ + k] = phi[t * K_ + k] + acc[k];
    }
}

// -----------------------------------------------------------------------------
// Kernel 2: main einsum.
// Grid (T/32, H, B); block 256 = 8 waves; wave w -> rowTile rt=w>>2 (0..1),
// colTile ct=w&3 (0..3). Each wave computes a 16x16 tile of out for 32 tokens
// x 64 outputs of head h:
//   P_k = Xtile(16x64,bf16) @ W[h,k](64x64,bf16)   -> 2x wmma_f32_16x16x32_bf16
//   acc[r] += scores[row_r, k] * P_k[r]            -> f32 VALU
// -----------------------------------------------------------------------------
__global__ __launch_bounds__(256) void moe_heads_kernel(
        const float* __restrict__ x,
        const unsigned short* __restrict__ wf,
        const float* __restrict__ scores,
        float* __restrict__ out) {
    const int tt = blockIdx.x, h = blockIdx.y, b = blockIdx.z;
    const int t0 = tt * TTILE;
    const int tid  = threadIdx.x;
    const int lane = tid & 31;
    const int wv   = tid >> 5;
    const int rt = wv >> 2;       // row tile (token group of 16)
    const int ct = wv & 3;        // col tile (output group of 16)
    const int laneHalf = (lane >= 16) ? 1 : 0;
    const int row = lane & 15;

    __shared__ unsigned short lds_x[TTILE * D_H];   // bf16 x tile, [t_local][i]  (4 KB)
    __shared__ float          lds_s[K_ * TTILE];    // scores transposed [k][t_local] (1 KB)

    // ---- stage x tile (f32 -> bf16) : 2048 elems, 8 per thread ----
    {
        const float* xb = x + ((size_t)(b * T_ + t0)) * D_IN + h * D_H;
#pragma unroll
        for (int j = 0; j < 8; ++j) {
            int e  = tid + j * 256;
            int tr = e >> 6;
            int ii = e & 63;
            lds_x[e] = f32_to_bf16_rne(xb[(size_t)tr * D_IN + ii]);
        }
    }
    // ---- stage scores transposed: tid = k*32 + t_local ----
    {
        int k  = tid >> 5;
        int tl = tid & 31;
        lds_s[k * TTILE + tl] = scores[((size_t)(b * T_ + t0 + tl)) * K_ + k];
    }
    __syncthreads();

    // ---- build A fragments (one per 32-wide K chunk of d_h=64) ----
    // 16-bit A 16x32 layout: dword p -> K = (p<4?0:16) + 8*laneHalf + 2*(p&3)
    union Frag { v16bf v; unsigned int u[8]; };
    Frag A0, A1;
    {
        const unsigned short* src = &lds_x[(rt * 16 + row) * D_H];
#pragma unroll
        for (int p = 0; p < 8; ++p) {
            int kbase = ((p < 4) ? 0 : 16) + laneHalf * 8 + 2 * (p & 3);
            A0.u[p] = *(const unsigned int*)&src[kbase];        // packed bf16 pair
            A1.u[p] = *(const unsigned int*)&src[32 + kbase];
        }
    }

    float accf[8];
#pragma unroll
    for (int r = 0; r < 8; ++r) accf[r] = 0.0f;

#pragma unroll
    for (int k = 0; k < K_; ++k) {
        // ---- load pre-swizzled B fragments: 2x (2x b128) coalesced loads ----
        Frag Bf0, Bf1;
        {
            const uint4* p0 = (const uint4*)(wf +
                ((size_t)(h * 64 + k * 8 + 0 * 4 + ct) * 512 + lane * 16));
            const uint4* p1 = (const uint4*)(wf +
                ((size_t)(h * 64 + k * 8 + 1 * 4 + ct) * 512 + lane * 16));
            uint4 q0 = p0[0], q1 = p0[1], q2 = p1[0], q3 = p1[1];
            Bf0.u[0] = q0.x; Bf0.u[1] = q0.y; Bf0.u[2] = q0.z; Bf0.u[3] = q0.w;
            Bf0.u[4] = q1.x; Bf0.u[5] = q1.y; Bf0.u[6] = q1.z; Bf0.u[7] = q1.w;
            Bf1.u[0] = q2.x; Bf1.u[1] = q2.y; Bf1.u[2] = q2.z; Bf1.u[3] = q2.w;
            Bf1.u[4] = q3.x; Bf1.u[5] = q3.y; Bf1.u[6] = q3.z; Bf1.u[7] = q3.w;
        }

        // ---- P_k = Xtile @ W[h,k] over d_h = 64 = 2 x K32 ----
        union { v8f v; float f[8]; } P;
#pragma unroll
        for (int r = 0; r < 8; ++r) P.f[r] = 0.0f;
        P.v = __builtin_amdgcn_wmma_f32_16x16x32_bf16(
                false, A0.v, false, Bf0.v, (short)0, P.v, false, false);
        P.v = __builtin_amdgcn_wmma_f32_16x16x32_bf16(
                false, A1.v, false, Bf1.v, (short)0, P.v, false, false);

        // ---- weighted accumulate: row r maps to t_local = rt*16 + laneHalf*8 + r
        const float* sp = &lds_s[k * TTILE + rt * 16 + laneHalf * 8];
#pragma unroll
        for (int r = 0; r < 8; ++r)
            accf[r] = fmaf(sp[r], P.f[r], accf[r]);
    }

    // ---- store: out[b][t][h*64 + o], f32 ----
    float* ob = out + ((size_t)(b * T_ + t0 + rt * 16 + laneHalf * 8)) * (H_ * D_O)
                    + h * D_O + ct * 16 + row;
#pragma unroll
    for (int r = 0; r < 8; ++r)
        ob[(size_t)r * (H_ * D_O)] = accf[r];
}

// -----------------------------------------------------------------------------
extern "C" void kernel_launch(void* const* d_in, const int* in_sizes, int n_in,
                              void* d_out, int out_size, void* d_ws, size_t ws_size,
                              hipStream_t stream) {
    (void)in_sizes; (void)n_in; (void)out_size; (void)ws_size;
    const float* x      = (const float*)d_in[0];
    const float* weight = (const float*)d_in[1];
    const float* diag   = (const float*)d_in[2];
    const float* phi    = (const float*)d_in[3];
    float* out = (float*)d_out;

    // workspace: [0, 1MB) bf16 swizzled weights; [1MB, 1.25MB) scores f32
    unsigned short* wf = (unsigned short*)d_ws;
    float* scores = (float*)((char*)d_ws + (size_t)H_ * K_ * D_H * D_O * sizeof(unsigned short));

    // 0) weight -> bf16 fragments (524288 elems)
    prep_weight_frag_kernel<<<(H_ * K_ * D_H * D_O) / 256, 256, 0, stream>>>(weight, wf);

    // 1) scores: one wave per token, 8 waves per block
    scores_kernel<<<(B_ * T_ * 32) / 256, 256, 0, stream>>>(x, diag, phi, scores);

    // 2) main einsum
    dim3 grid(T_ / TTILE, H_, B_);
    moe_heads_kernel<<<grid, 256, 0, stream>>>(x, wf, scores, out);
}
